// GAT_2748779070163
// MI455X (gfx1250) — matrix operand order repaired and checked by reading, not compiled, not run on hardware
//
#include <hip/hip_runtime.h>
#include <hip/hip_bf16.h>
#include <stdint.h>

// ---------------- problem constants ----------------
#define NN     10000     // nodes
#define NP     10048     // nodes padded to multiple of 64 (16 rows x 4 waves)
#define EE     160000    // raw edges
#define ET     170000    // edges + self loops
#define DIN    54
#define KP1    64        // K padded for layer-1 GEMM
#define DH     64
#define NH     8
#define HID    512       // NH*DH
#define DOUT   4
#define NEG    0.2f

// WMMA fragment types (gfx1250, wave32)
typedef __attribute__((ext_vector_type(16))) _Float16 v16h;
typedef __attribute__((ext_vector_type(8)))  _Float16 v8h;
typedef __attribute__((ext_vector_type(8)))  float    v8f;

// ---- monotone int encoding of floats so atomicMax(int) == float max ----
__device__ __forceinline__ int   fenc(float f) { int i = __float_as_int(f); return i < 0 ? (i ^ 0x7fffffff) : i; }
__device__ __forceinline__ float fdec(int i)   { return __int_as_float(i < 0 ? (i ^ 0x7fffffff) : i); }

__device__ __forceinline__ float lrelu(float v) { return v > 0.f ? v : NEG * v; }

// assemble v16h from two contiguous v8h halves
__device__ __forceinline__ v16h cat16(v8h lo, v8h hi) {
    return __builtin_shufflevector(lo, hi, 0,1,2,3,4,5,6,7,8,9,10,11,12,13,14,15);
}

// ---------------- init kernels ----------------
__global__ void fill_f32(float* p, int n, float v) {
    int i = blockIdx.x * blockDim.x + threadIdx.x;
    if (i < n) p[i] = v;
}
__global__ void fill_i32(int* p, int n, int v) {
    int i = blockIdx.x * blockDim.x + threadIdx.x;
    if (i < n) p[i] = v;
}
__global__ void init_out_bias(float* out, const float* __restrict__ bias2) {
    int i = blockIdx.x * blockDim.x + threadIdx.x;
    if (i < NN * DOUT) out[i] = bias2[i & 3];
}

// ---------------- one-time operand packing (zero-padded f16, L2-resident) ----------------
// x [NN,54] f32 -> Ap1 [NP,64] f16
__global__ void pack_x_k(const float* __restrict__ x, _Float16* __restrict__ Ap1) {
    int t = blockIdx.x * blockDim.x + threadIdx.x;
    if (t >= NP * KP1) return;
    const int m = t >> 6, k = t & 63;
    float v = 0.f;
    if (m < NN && k < DIN) v = x[m * DIN + k];
    Ap1[t] = (_Float16)v;
}
// Wl1|Wr1 [54,512] f32 -> Bp1 [1024,64] f16  (n-major: row n holds K=0..63)
__global__ void pack_w1_k(const float* __restrict__ Wl, const float* __restrict__ Wr,
                          _Float16* __restrict__ Bp1) {
    int t = blockIdx.x * blockDim.x + threadIdx.x;
    if (t >= 1024 * KP1) return;
    const int n = t >> 6, k = t & 63;
    float v = 0.f;
    if (k < DIN) v = (n < HID) ? Wl[k * HID + n] : Wr[k * HID + (n - HID)];
    Bp1[t] = (_Float16)v;
}
// Wl2|Wr2 [512,4] f32 -> Bp2 [16,512] f16 (n-major, cols 8..15 zero)
__global__ void pack_w2_k(const float* __restrict__ Wl, const float* __restrict__ Wr,
                          _Float16* __restrict__ Bp2) {
    int t = blockIdx.x * blockDim.x + threadIdx.x;
    if (t >= 16 * HID) return;
    const int n = t >> 9, k = t & 511;
    float v = 0.f;
    if (n < DOUT)          v = Wl[k * DOUT + n];
    else if (n < 2 * DOUT) v = Wr[k * DOUT + (n - DOUT)];
    Bp2[t] = (_Float16)v;
}
// concatenated bias vectors: bias1c[1024] = bl1|br1, bias2c[16] = bl2|br2|0
__global__ void pack_bias_k(const float* __restrict__ bl1, const float* __restrict__ br1,
                            const float* __restrict__ bl2, const float* __restrict__ br2,
                            float* __restrict__ b1c, float* __restrict__ b2c) {
    int t = blockIdx.x * blockDim.x + threadIdx.x;
    if (t < 1024) b1c[t] = (t < HID) ? bl1[t] : br1[t - HID];
    if (t < 16) {
        float v = 0.f;
        if (t < DOUT)          v = bl2[t];
        else if (t < 2 * DOUT) v = br2[t - DOUT];
        b2c[t] = v;
    }
}

// ---------------- layer-1 GEMM: Ap1[NP,64] x Bp1 -> xlr1 [NP,1024] ----------------
// Branch-free: every fragment = two global b128 loads. 4 waves/block, 16x16 tile/wave.
// grid = (NP/16, 1024/64), block = 128
__global__ void __launch_bounds__(128) gemm1_wmma(
    const _Float16* __restrict__ Ap1, const _Float16* __restrict__ Bp1,
    const float* __restrict__ b1c, float* __restrict__ xlr)
{
    const int lane = threadIdx.x & 31;
    const int wave = threadIdx.x >> 5;
    const int row0 = blockIdx.x * 16;
    const int n0   = blockIdx.y * 64 + wave * 16;
    const int half = lane >> 4;       // 0: lanes 0-15, 1: lanes 16-31
    const int l15  = lane & 15;

    const int m = row0 + l15;         // always < NP (padded)
    const int n = n0 + l15;
    const _Float16* arow = Ap1 + (size_t)m * KP1;
    const _Float16* brow = Bp1 + (size_t)n * KP1;

    v8f acc = {};
    #pragma unroll
    for (int kb = 0; kb < KP1; kb += 32) {
        // A 16x32 f16: elems 0..7 -> K = kb + half*8 + 0..7 ; 8..15 -> +16
        v16h a = cat16(*(const v8h*)(arow + kb + half * 8),
                       *(const v8h*)(arow + kb + half * 8 + 16));
        // B 32x16 f16: elems 0..7 -> K = kb + half*8 + 0..7 ; 8..15 -> +16 (n-major pack)
        v16h b = cat16(*(const v8h*)(brow + kb + half * 8),
                       *(const v8h*)(brow + kb + half * 8 + 16));
        acc = __builtin_amdgcn_wmma_f32_16x16x32_f16(false, a, false, b, (short)0, acc, false, false);
    }
    const float bias = b1c[n];
    #pragma unroll
    for (int r = 0; r < 8; ++r) {
        const int mr = row0 + half * 8 + r;
        xlr[(size_t)mr * 1024 + n] = acc[r] + bias;   // padded store, no guard
    }
}

// ---------------- layer-2 GEMM: h1h[NP,512] x Bp2[16,512] -> xlr2 [NP,16] ----------------
// one 16-row tile per wave; grid = NP/64, block = 128; fully unrolled K (16 WMMAs)
__global__ void __launch_bounds__(128) gemm2_wmma(
    const _Float16* __restrict__ h1h, const _Float16* __restrict__ Bp2,
    const float* __restrict__ b2c, float* __restrict__ xlr2)
{
    const int lane = threadIdx.x & 31;
    const int wave = threadIdx.x >> 5;
    const int row0 = (blockIdx.x * 4 + wave) * 16;   // < NP always
    const int half = lane >> 4;
    const int l15  = lane & 15;

    const int m = row0 + l15;
    const _Float16* arow = h1h + (size_t)m * HID;
    const _Float16* brow = Bp2 + (size_t)l15 * HID;

    v8f acc = {};
    #pragma unroll
    for (int kb = 0; kb < HID; kb += 32) {
        v16h a = cat16(*(const v8h*)(arow + kb + half * 8),
                       *(const v8h*)(arow + kb + half * 8 + 16));
        v16h b = cat16(*(const v8h*)(brow + kb + half * 8),
                       *(const v8h*)(brow + kb + half * 8 + 16));
        acc = __builtin_amdgcn_wmma_f32_16x16x32_f16(false, a, false, b, (short)0, acc, false, false);
    }
    const float bias = b2c[l15];
    #pragma unroll
    for (int r = 0; r < 8; ++r) {
        const int mr = row0 + half * 8 + r;
        xlr2[(size_t)mr * 16 + l15] = acc[r] + bias;  // padded 16-wide store, no guard
    }
}

// ---------------- edge helpers ----------------
__device__ __forceinline__ void edge_sd(const int* __restrict__ ei, int e, int& s, int& d) {
    if (e < EE) { s = ei[e]; d = ei[EE + e]; }
    else        { s = e - EE; d = e - EE; }   // self loop
}

// ---------------- layer-1 edge passes ----------------
__global__ void score1_k(const float* __restrict__ xlr, const int* __restrict__ ei,
                         const float* __restrict__ att1, float* __restrict__ e1,
                         int* __restrict__ m1)
{
    int t = blockIdx.x * blockDim.x + threadIdx.x;
    if (t >= ET * NH) return;
    const int e = t >> 3, h = t & 7;
    int s, d; edge_sd(ei, e, s, d);
    const float4* xl = (const float4*)(xlr + (size_t)s * 1024 + h * DH);
    const float4* xr = (const float4*)(xlr + (size_t)d * 1024 + HID + h * DH);
    const float4* at = (const float4*)(att1 + h * DH);
    float acc = 0.f;
    #pragma unroll
    for (int q = 0; q < DH / 4; ++q) {
        const float4 a = xl[q], b = xr[q], w = at[q];
        acc += w.x * lrelu(a.x + b.x);
        acc += w.y * lrelu(a.y + b.y);
        acc += w.z * lrelu(a.z + b.z);
        acc += w.w * lrelu(a.w + b.w);
    }
    e1[t] = acc;
    atomicMax(m1 + d * NH + h, fenc(acc));
}

__global__ void prob1_k(const float* __restrict__ e1, const int* __restrict__ ei,
                        const int* __restrict__ m1, float* __restrict__ p1,
                        float* __restrict__ s1)
{
    int t = blockIdx.x * blockDim.x + threadIdx.x;
    if (t >= ET * NH) return;
    const int e = t >> 3, h = t & 7;
    int s, d; edge_sd(ei, e, s, d);
    const float p = __expf(e1[t] - fdec(m1[d * NH + h]));
    p1[t] = p;
    atomicAdd(s1 + d * NH + h, p);
}

__global__ void agg1_k(const float* __restrict__ xlr, const int* __restrict__ ei,
                       const float* __restrict__ p1, const float* __restrict__ s1,
                       float* __restrict__ agg1)
{
    int t = blockIdx.x * blockDim.x + threadIdx.x;
    if (t >= ET * (HID / 4)) return;
    const int e = t >> 7, q = t & 127;
    const int c = q * 4, h = c >> 6;
    int s, d; edge_sd(ei, e, s, d);
    const float alpha = p1[e * NH + h] / (s1[d * NH + h] + 1e-16f);
    const float4 v = *(const float4*)(xlr + (size_t)s * 1024 + c);
    float* o = agg1 + (size_t)d * HID + c;
    atomicAdd(o + 0, alpha * v.x);
    atomicAdd(o + 1, alpha * v.y);
    atomicAdd(o + 2, alpha * v.z);
    atomicAdd(o + 3, alpha * v.w);
}

// fused bias + ReLU + f16 pack for layer-2 GEMM operand (covers padded rows too)
__global__ void relu_bias_pack_k(const float* __restrict__ agg1,
                                 const float* __restrict__ bias1,
                                 _Float16* __restrict__ h1h)
{
    int t = blockIdx.x * blockDim.x + threadIdx.x;
    if (t >= NP * HID) return;
    h1h[t] = (_Float16)fmaxf(agg1[t] + bias1[t & (HID - 1)], 0.f);
}

// ---------------- layer-2 edge passes (H=1, D=4; xlr2 stride 16) ----------------
__global__ void score2_k(const float* __restrict__ xlr2, const int* __restrict__ ei,
                         const float* __restrict__ att2, float* __restrict__ e2,
                         int* __restrict__ m2)
{
    int e = blockIdx.x * blockDim.x + threadIdx.x;
    if (e >= ET) return;
    int s, d; edge_sd(ei, e, s, d);
    const float4 xl = *(const float4*)(xlr2 + (size_t)s * 16);
    const float4 xr = *(const float4*)(xlr2 + (size_t)d * 16 + 4);
    float acc = att2[0] * lrelu(xl.x + xr.x)
              + att2[1] * lrelu(xl.y + xr.y)
              + att2[2] * lrelu(xl.z + xr.z)
              + att2[3] * lrelu(xl.w + xr.w);
    e2[e] = acc;
    atomicMax(m2 + d, fenc(acc));
}

__global__ void prob2_k(const float* __restrict__ e2, const int* __restrict__ ei,
                        const int* __restrict__ m2, float* __restrict__ p2,
                        float* __restrict__ s2)
{
    int e = blockIdx.x * blockDim.x + threadIdx.x;
    if (e >= ET) return;
    int s, d; edge_sd(ei, e, s, d);
    const float p = __expf(e2[e] - fdec(m2[d]));
    p2[e] = p;
    atomicAdd(s2 + d, p);
}

__global__ void agg2_k(const float* __restrict__ xlr2, const int* __restrict__ ei,
                       const float* __restrict__ p2, const float* __restrict__ s2,
                       float* __restrict__ out, float* __restrict__ alpha_out)
{
    int e = blockIdx.x * blockDim.x + threadIdx.x;
    if (e >= ET) return;
    int s, d; edge_sd(ei, e, s, d);
    const float alpha = p2[e] / (s2[d] + 1e-16f);
    alpha_out[e] = alpha;
    const float4 xl = *(const float4*)(xlr2 + (size_t)s * 16);
    float* o = out + (size_t)d * DOUT;
    atomicAdd(o + 0, alpha * xl.x);
    atomicAdd(o + 1, alpha * xl.y);
    atomicAdd(o + 2, alpha * xl.z);
    atomicAdd(o + 3, alpha * xl.w);
}

__global__ void ei_out_k(const int* __restrict__ ei, float* __restrict__ ei_out)
{
    int e = blockIdx.x * blockDim.x + threadIdx.x;
    if (e >= ET) return;
    int s, d; edge_sd(ei, e, s, d);
    ei_out[e]      = (float)s;
    ei_out[ET + e] = (float)d;
}

// ---------------- host launch ----------------
extern "C" void kernel_launch(void* const* d_in, const int* in_sizes, int n_in,
                              void* d_out, int out_size, void* d_ws, size_t ws_size,
                              hipStream_t stream) {
    const float* x     = (const float*)d_in[0];
    const int*   ei    = (const int*)d_in[1];
    const float* Wl1   = (const float*)d_in[2];
    const float* bl1   = (const float*)d_in[3];
    const float* Wr1   = (const float*)d_in[4];
    const float* br1   = (const float*)d_in[5];
    const float* att1  = (const float*)d_in[6];
    const float* bias1 = (const float*)d_in[7];
    const float* Wl2   = (const float*)d_in[8];
    const float* bl2   = (const float*)d_in[9];
    const float* Wr2   = (const float*)d_in[10];
    const float* br2   = (const float*)d_in[11];
    const float* att2  = (const float*)d_in[12];
    const float* bias2 = (const float*)d_in[13];

    // workspace carve-up (float units; every block is a multiple of 16 floats)
    float* ws = (float*)d_ws;
    size_t off = 0;
    float*    xlr1 = ws + off; off += (size_t)NP * 1024;          // f32 [NP,1024]
    float*    e1   = ws + off; off += (size_t)ET * NH;
    float*    p1   = ws + off; off += (size_t)ET * NH;
    int*      m1   = (int*)(ws + off); off += NN * NH;
    float*    s1   = ws + off; off += NN * NH;
    float*    agg1 = ws + off; off += (size_t)NP * HID;           // f32 [NP,512]
    float*    xlr2 = ws + off; off += (size_t)NP * 16;            // f32 [NP,16]
    float*    e2   = ws + off; off += ET;
    float*    p2   = ws + off; off += ET + 64 - (ET & 63);        // pad to a multiple of 64
    int*      m2   = (int*)(ws + off); off += NN;
    float*    s2   = ws + off; off += NN;
    float*    b1c  = ws + off; off += 1024;
    float*    b2c  = ws + off; off += 64;
    _Float16* Ap1  = (_Float16*)(ws + off); off += (size_t)NP * KP1 / 2;   // f16 [NP,64]
    _Float16* Bp1  = (_Float16*)(ws + off); off += 1024 * KP1 / 2;         // f16 [1024,64]
    _Float16* Bp2  = (_Float16*)(ws + off); off += 16 * HID / 2;           // f16 [16,512]
    _Float16* h1h  = (_Float16*)(ws + off); off += (size_t)NP * HID / 2;   // f16 [NP,512]

    // d_out tuple layout: out [N*4] | ei_sl [2*ET] | alpha [ET]
    float* out_p   = (float*)d_out;
    float* eisl_p  = out_p + NN * DOUT;
    float* alpha_p = eisl_p + 2 * ET;

    const int ENC_NEG_INF = (int)0x807FFFFFu;       // fenc(-inf)
    const int B = 256;

    // --- init + operand packing ---
    fill_i32<<<(NN * NH + B - 1) / B, B, 0, stream>>>(m1, NN * NH, ENC_NEG_INF);
    fill_f32<<<(NN * NH + B - 1) / B, B, 0, stream>>>(s1, NN * NH, 0.f);
    fill_f32<<<(NP * HID + B - 1) / B, B, 0, stream>>>(agg1, NP * HID, 0.f);
    fill_i32<<<(NN + B - 1) / B, B, 0, stream>>>(m2, NN, ENC_NEG_INF);
    fill_f32<<<(NN + B - 1) / B, B, 0, stream>>>(s2, NN, 0.f);
    init_out_bias<<<(NN * DOUT + B - 1) / B, B, 0, stream>>>(out_p, bias2);
    pack_x_k<<<(NP * KP1 + B - 1) / B, B, 0, stream>>>(x, Ap1);
    pack_w1_k<<<(1024 * KP1 + B - 1) / B, B, 0, stream>>>(Wl1, Wr1, Bp1);
    pack_w2_k<<<(16 * HID + B - 1) / B, B, 0, stream>>>(Wl2, Wr2, Bp2);
    pack_bias_k<<<(1024 + B - 1) / B, B, 0, stream>>>(bl1, br1, bl2, br2, b1c, b2c);

    // --- layer 1 ---
    gemm1_wmma<<<dim3(NP / 16, 1024 / 64), 128, 0, stream>>>(Ap1, Bp1, b1c, xlr1);
    score1_k<<<(ET * NH + B - 1) / B, B, 0, stream>>>(xlr1, ei, att1, e1, m1);
    prob1_k<<<(ET * NH + B - 1) / B, B, 0, stream>>>(e1, ei, m1, p1, s1);
    agg1_k<<<(ET * (HID / 4) + B - 1) / B, B, 0, stream>>>(xlr1, ei, p1, s1, agg1);
    relu_bias_pack_k<<<(NP * HID + B - 1) / B, B, 0, stream>>>(agg1, bias1, h1h);

    // --- layer 2 ---
    gemm2_wmma<<<NP / 64, 128, 0, stream>>>(h1h, Bp2, b2c, xlr2);
    score2_k<<<(ET + B - 1) / B, B, 0, stream>>>(xlr2, ei, att2, e2, m2);
    prob2_k<<<(ET + B - 1) / B, B, 0, stream>>>(e2, ei, m2, p2, s2);
    ei_out_k<<<(ET + B - 1) / B, B, 0, stream>>>(ei, eisl_p);
    agg2_k<<<(ET + B - 1) / B, B, 0, stream>>>(xlr2, ei, p2, s2, out_p, alpha_p);
}
